// HardQuadTripletSOSRLoss_29446295781454
// MI455X (gfx1250) — matrix operand order repaired and checked by reading, not compiled, not run on hardware
//
#include <hip/hip_runtime.h>
#include <hip/hip_bf16.h>
#include <math.h>

// ---------------- problem constants (from reference) ----------------
#define BB 4
#define NN 1024
#define CC 128
#define HH 64
#define WW 64
#define HWW 4096
#define GS 8.0f
#define NUM_NEG 8
#define SOS_NEG 8
#define MARGIN 1.0f

typedef __attribute__((ext_vector_type(16))) __bf16 v16bf;
typedef __attribute__((ext_vector_type(8)))  float  v8f;
typedef __attribute__((ext_vector_type(8)))  unsigned int v8u;

// float -> bf16 (round to nearest even), stored as ushort
__device__ inline unsigned short f2bf(float f) {
    unsigned u = __builtin_bit_cast(unsigned, f);
    unsigned r = (u + 0x7FFFu + ((u >> 16) & 1u)) >> 16;
    return (unsigned short)r;
}

__device__ inline float wave_sum32(float v) {
    v += __shfl_xor(v, 16, 32);
    v += __shfl_xor(v, 8, 32);
    v += __shfl_xor(v, 4, 32);
    v += __shfl_xor(v, 2, 32);
    v += __shfl_xor(v, 1, 32);
    return v;
}

__device__ inline int iclamp(int v, int lo, int hi) {
    return v < lo ? lo : (v > hi ? hi : v);
}

// 4 nearest grid-cell centers of point (x,y); 3x3 in-range stencil, pick 4 smallest
__device__ inline void nearest4(float x, float y, int* out4) {
    int ixn = iclamp((int)floorf(x * 0.125f), 0, WW - 1);
    int iyn = iclamp((int)floorf(y * 0.125f), 0, HH - 1);
    float cd[9];
    int cid[9];
    int cnt = 0;
    for (int dy = -1; dy <= 1; ++dy)
        for (int dx = -1; dx <= 1; ++dx) {
            int ix = ixn + dx, iy = iyn + dy;
            if (ix < 0 || ix >= WW || iy < 0 || iy >= HH) continue;
            float cx = (ix + 0.5f) * GS, cy = (iy + 0.5f) * GS;
            float ddx = x - cx, ddy = y - cy;
            cd[cnt] = ddx * ddx + ddy * ddy;
            cid[cnt] = iy * WW + ix;
            cnt++;
        }
    for (int k = 0; k < 4; ++k) {
        int best = 0;
        float bv = cd[0];
        for (int i = 1; i < cnt; ++i)
            if (cd[i] < bv) { bv = cd[i]; best = i; }
        out4[k] = cid[best];
        cd[best] = 3e38f;
    }
}

// ---------------- kernel 0: zero accumulators ----------------
__global__ void init_acc_kernel(float* acc) {
    if (threadIdx.x < 2) acc[threadIdx.x] = 0.0f;
}

// ---------------- kernel 1: desc2 (B,C,H,W) -> bf16 (B,HW,C) ----------------
__global__ __launch_bounds__(128) void conv_d2_kernel(const float* __restrict__ desc2,
                                                      unsigned short* __restrict__ d2bf) {
    int p = blockIdx.x;           // b*HW + hw
    int b = p >> 12;
    int hw = p & (HWW - 1);
    int c = threadIdx.x;
    float v = desc2[(((size_t)b * CC + c) << 12) + hw];
    d2bf[((size_t)p << 7) + c] = f2bf(v);
}

// ---------------- kernel 2: bilinear sample + normalize + pos_sim + bf16 + excl ids ----
__global__ __launch_bounds__(32) void prep_kernel(const float* __restrict__ kp1,
                                                  const float* __restrict__ w_kp1,
                                                  const float* __restrict__ kp1_desc,
                                                  const float* __restrict__ desc2,
                                                  const float* __restrict__ homo,
                                                  float* __restrict__ wdesc,
                                                  unsigned short* __restrict__ abf,
                                                  unsigned short* __restrict__ wbf,
                                                  float* __restrict__ pos_sim,
                                                  int* __restrict__ excl) {
    int p = blockIdx.x;           // b*N + n
    int b = p >> 10;
    int lane = threadIdx.x;

    float wx = w_kp1[p * 2 + 0], wy = w_kp1[p * 2 + 1];
    float gx = wx * 0.125f - 0.5f, gy = wy * 0.125f - 0.5f;
    float x0f = floorf(gx), y0f = floorf(gy);
    float fx = gx - x0f, fy = gy - y0f;
    int x0 = iclamp((int)x0f, 0, WW - 1);
    int x1 = iclamp(x0 + 1, 0, WW - 1);
    int y0 = iclamp((int)y0f, 0, HH - 1);
    int y1 = iclamp(y0 + 1, 0, HH - 1);
    float w00 = (1.f - fx) * (1.f - fy), w10 = fx * (1.f - fy);
    float w01 = (1.f - fx) * fy, w11 = fx * fy;

    float out[4];
    float ss = 0.f;
#pragma unroll
    for (int j = 0; j < 4; ++j) {
        int c = lane + 32 * j;
        const float* base = desc2 + (((size_t)b * CC + c) << 12);
        float v = w00 * base[y0 * WW + x0] + w10 * base[y0 * WW + x1] +
                  w01 * base[y1 * WW + x0] + w11 * base[y1 * WW + x1];
        out[j] = v;
        ss += v * v;
    }
    ss = wave_sum32(ss);
    float scale = 1.0f / (sqrtf(ss) + 1e-8f);

    float dot = 0.f;
#pragma unroll
    for (int j = 0; j < 4; ++j) {
        int c = lane + 32 * j;
        float v = out[j] * scale;
        wdesc[(size_t)p * CC + c] = v;
        wbf[(size_t)p * CC + c] = f2bf(v);
        float a = kp1_desc[(size_t)p * CC + c];
        abf[(size_t)p * CC + c] = f2bf(a);
        dot += a * v;
    }
    dot = wave_sum32(dot);

    if (lane == 0) {
        pos_sim[p] = 2.0f - 2.0f * dot;
        // exclusion ids: 4 nearest cells of kp1 -> warp -> 4 nearest cells each
        float px = kp1[p * 2 + 0], py = kp1[p * 2 + 1];
        int cells[4];
        nearest4(px, py, cells);
        float h[9];
        for (int i = 0; i < 9; ++i) h[i] = homo[b * 9 + i];
        for (int c4 = 0; c4 < 4; ++c4) {
            int id = cells[c4];
            float cx = ((id & (WW - 1)) + 0.5f) * GS;
            float cy = ((id >> 6) + 0.5f) * GS;
            float q0 = h[0] * cx + h[1] * cy + h[2];
            float q1 = h[3] * cx + h[4] * cy + h[5];
            float q2 = h[6] * cx + h[7] * cy + h[8];
            float wxc = q0 / (q2 + 1e-8f);
            float wyc = q1 / (q2 + 1e-8f);
            int c2[4];
            nearest4(wxc, wyc, c2);
            for (int j = 0; j < 4; ++j)
                excl[(size_t)p * 16 + c4 * 4 + j] = c2[j];
        }
    }
}

// ---- helper: load A fragment (16x32 bf16, MxK) for this lane ----
__device__ inline v16bf load_a_frag(const unsigned short* __restrict__ A, int row0,
                                    int m16, int half, int kk) {
    v8u ai;
#pragma unroll
    for (int j = 0; j < 8; ++j) {
        int kb = 32 * kk + ((j < 4) ? 2 * j : 16 + 2 * (j - 4)) + 8 * half;
        ai[j] = *(const unsigned*)(A + (size_t)(row0 + m16) * CC + kb);
    }
    return __builtin_bit_cast(v16bf, ai);
}

// ---- helper: load B fragment (32x16 bf16, KxN); B[k][n] = Bm[n*C + k] ----
__device__ inline v16bf load_b_frag(const unsigned short* __restrict__ Bm, int n0,
                                    int m16, int half, int kk) {
    v8u bi;
#pragma unroll
    for (int j = 0; j < 8; ++j) {
        int k = 32 * kk + ((j < 4) ? 2 * j : 16 + 2 * (j - 4)) + 8 * half;
        bi[j] = *(const unsigned*)(Bm + (size_t)(n0 + m16) * CC + k);
    }
    return __builtin_bit_cast(v16bf, bi);
}

// ---------------- kernel 3: desc_sim WMMA + masked top-8 + fos partial ------------
__global__ __launch_bounds__(32) void fos_kernel(const unsigned short* __restrict__ abf,
                                                 const unsigned short* __restrict__ d2bf,
                                                 const float* __restrict__ pos_sim,
                                                 const int* __restrict__ excl,
                                                 float* __restrict__ fosAcc) {
    int blk = blockIdx.x;         // b*64 + rowblock
    int b = blk >> 6;
    int row0 = (blk & 63) * 16;
    int lane = threadIdx.x;
    int half = lane >> 4, m16 = lane & 15;

    const unsigned short* A = abf + (size_t)b * NN * CC;
    const unsigned short* Bm = d2bf + (size_t)b * HWW * CC;

    v16bf afrag[4];
#pragma unroll
    for (int kk = 0; kk < 4; ++kk) afrag[kk] = load_a_frag(A, row0, m16, half, kk);

    // exclusion ids held in registers (lanes 0..15, one row each; unrolled const idx)
    int eid[16];
    {
        int r = (lane < 16) ? lane : 0;
        const int* ep = excl + ((size_t)b * NN + row0 + r) * 16;
#pragma unroll
        for (int e = 0; e < 16; ++e) eid[e] = ep[e];
    }
    float posv = (lane < 16) ? pos_sim[b * NN + row0 + lane] : 0.f;

    __shared__ float tile[16][16];

    float t[8];
#pragma unroll
    for (int k = 0; k < 8; ++k) t[k] = 3e38f;

    const int NT = HWW / 16;
    // software pipeline: preload tile 0 fragments
    v16bf bcur[4];
#pragma unroll
    for (int kk = 0; kk < 4; ++kk) bcur[kk] = load_b_frag(Bm, 0, m16, half, kk);

    for (int nt = 0; nt < NT; ++nt) {
        int n0 = nt * 16;
        int n1 = (nt + 1 < NT) ? (nt + 1) * 16 : n0;
        // prefetch 2 tiles ahead
        if (nt + 2 < NT)
            __builtin_prefetch(Bm + (size_t)(n0 + 32 + m16) * CC, 0, 0);
        // issue next tile's loads (overlap with WMMA + scan below)
        v16bf bnext[4];
#pragma unroll
        for (int kk = 0; kk < 4; ++kk) bnext[kk] = load_b_frag(Bm, n1, m16, half, kk);

        v8f acc = {0.f, 0.f, 0.f, 0.f, 0.f, 0.f, 0.f, 0.f};
#pragma unroll
        for (int kk = 0; kk < 4; ++kk) {
            acc = __builtin_amdgcn_wmma_f32_16x16x32_bf16(
                false, afrag[kk], false, bcur[kk], (short)0, acc, false, false);
        }
#pragma unroll
        for (int j = 0; j < 8; ++j)
            tile[j + 8 * half][m16] = 2.0f - 2.0f * acc[j];
        __syncthreads();
        if (lane < 16) {
            // branchless 16-bit exclusion mask for this column tile
            unsigned mask = 0u;
#pragma unroll
            for (int e = 0; e < 16; ++e) {
                unsigned d = (unsigned)(eid[e] - n0);
                mask |= (d < 16u) ? (1u << d) : 0u;
            }
#pragma unroll
            for (int jj = 0; jj < 16; ++jj) {
                float v = tile[lane][jj] + (((mask >> jj) & 1u) ? 5.0f : 0.0f);
                if (v < t[7]) {
                    t[7] = v;
#pragma unroll
                    for (int q = 7; q >= 1; --q)
                        if (t[q] < t[q - 1]) { float tm = t[q]; t[q] = t[q - 1]; t[q - 1] = tm; }
                }
            }
        }
        __syncthreads();
#pragma unroll
        for (int kk = 0; kk < 4; ++kk) bcur[kk] = bnext[kk];
    }

    if (lane < 16) {
        float s = 0.f;
#pragma unroll
        for (int k = 0; k < 8; ++k) {
            float d = posv - t[k] + MARGIN;
            d = fmaxf(d, 0.f);
            s += d * d;
        }
        atomicAdd(fosAcc, s);
    }
}

// ---------------- kernel 4: self-sim WMMA + radius mask -> top-8 indices ----------
__global__ __launch_bounds__(32) void topk_ids_kernel(const unsigned short* __restrict__ dbf,
                                                      const float* __restrict__ pts,
                                                      int* __restrict__ out_ids) {
    int blk = blockIdx.x;         // b*64 + rowblock
    int b = blk >> 6;
    int row0 = (blk & 63) * 16;
    int lane = threadIdx.x;
    int half = lane >> 4, m16 = lane & 15;

    const float R = GS * 1.41421356237f + 0.1f;
    const float R2 = R * R;

    const unsigned short* A = dbf + (size_t)b * NN * CC;

    v16bf afrag[4];
#pragma unroll
    for (int kk = 0; kk < 4; ++kk) afrag[kk] = load_a_frag(A, row0, m16, half, kk);

    float px = 0.f, py = 0.f;
    if (lane < 16) {
        px = pts[((size_t)b * NN + row0 + lane) * 2 + 0];
        py = pts[((size_t)b * NN + row0 + lane) * 2 + 1];
    }

    __shared__ float tile[16][16];
    __shared__ float cpx[16], cpy[16];

    float t[8];
    int ti[8];
#pragma unroll
    for (int k = 0; k < 8; ++k) { t[k] = 3e38f; ti[k] = 0; }

    const int NT = NN / 16;
    v16bf bcur[4];
#pragma unroll
    for (int kk = 0; kk < 4; ++kk) bcur[kk] = load_b_frag(A, 0, m16, half, kk);

    for (int nt = 0; nt < NT; ++nt) {
        int n0 = nt * 16;
        int n1 = (nt + 1 < NT) ? (nt + 1) * 16 : n0;
        if (lane < 16) {
            cpx[lane] = pts[((size_t)b * NN + n0 + lane) * 2 + 0];
            cpy[lane] = pts[((size_t)b * NN + n0 + lane) * 2 + 1];
        }
        v16bf bnext[4];
#pragma unroll
        for (int kk = 0; kk < 4; ++kk) bnext[kk] = load_b_frag(A, n1, m16, half, kk);

        v8f acc = {0.f, 0.f, 0.f, 0.f, 0.f, 0.f, 0.f, 0.f};
#pragma unroll
        for (int kk = 0; kk < 4; ++kk) {
            acc = __builtin_amdgcn_wmma_f32_16x16x32_bf16(
                false, afrag[kk], false, bcur[kk], (short)0, acc, false, false);
        }
#pragma unroll
        for (int j = 0; j < 8; ++j)
            tile[j + 8 * half][m16] = 2.0f - 2.0f * acc[j];
        __syncthreads();
        if (lane < 16) {
#pragma unroll
            for (int jj = 0; jj < 16; ++jj) {
                float dx = px - cpx[jj], dy = py - cpy[jj];
                float v = tile[lane][jj] +
                          ((dx * dx + dy * dy <= R2) ? 5.0f : 0.0f);
                if (v < t[7]) {
                    t[7] = v;
                    ti[7] = n0 + jj;
#pragma unroll
                    for (int q = 7; q >= 1; --q)
                        if (t[q] < t[q - 1]) {
                            float tm = t[q]; t[q] = t[q - 1]; t[q - 1] = tm;
                            int im = ti[q]; ti[q] = ti[q - 1]; ti[q - 1] = im;
                        }
                }
            }
        }
        __syncthreads();
#pragma unroll
        for (int kk = 0; kk < 4; ++kk) bcur[kk] = bnext[kk];
    }

    if (lane < 16) {
#pragma unroll
        for (int k = 0; k < 8; ++k)
            out_ids[((size_t)b * NN + row0 + lane) * 8 + k] = ti[k];
    }
}

// ---------------- kernel 5: sos term ----------------
__global__ __launch_bounds__(32) void sos_kernel(const float* __restrict__ kp1_desc,
                                                 const float* __restrict__ wdesc,
                                                 const int* __restrict__ idsA,
                                                 const int* __restrict__ idsW,
                                                 float* __restrict__ sosAcc) {
    int p = blockIdx.x;           // b*N + n
    int b = p >> 10;
    int lane = threadIdx.x;

    float a[4], w[4];
#pragma unroll
    for (int j = 0; j < 4; ++j) {
        int c = lane + 32 * j;
        a[j] = kp1_desc[(size_t)p * CC + c];
        w[j] = wdesc[(size_t)p * CC + c];
    }
    float acc = 0.f;
    for (int k = 0; k < 8; ++k) {
        int ia = idsA[(size_t)p * 8 + k];
        int iw = idsW[(size_t)p * 8 + k];
        const float* na = kp1_desc + ((size_t)b * NN + ia) * CC;
        const float* nw = wdesc + ((size_t)b * NN + iw) * CC;
        float da = 0.f, dw = 0.f;
#pragma unroll
        for (int j = 0; j < 4; ++j) {
            int c = lane + 32 * j;
            da += a[j] * na[c];
            dw += w[j] * nw[c];
        }
        da = wave_sum32(da);
        dw = wave_sum32(dw);
        float s = 2.0f * (dw - da);  // (2-2da) - (2-2dw)
        acc += s * s;
    }
    if (lane == 0) atomicAdd(sosAcc, sqrtf(acc));
}

// ---------------- kernel 6: finalize ----------------
__global__ void finalize_kernel(const float* __restrict__ acc, float* __restrict__ out) {
    out[0] = acc[0] / (float)(BB * NN * NUM_NEG) + acc[1] / (float)(BB * NN);
}

extern "C" void kernel_launch(void* const* d_in, const int* in_sizes, int n_in,
                              void* d_out, int out_size, void* d_ws, size_t ws_size,
                              hipStream_t stream) {
    (void)in_sizes; (void)n_in; (void)out_size; (void)ws_size;
    const float* kp1 = (const float*)d_in[0];
    const float* w_kp1 = (const float*)d_in[1];
    const float* kp1_desc = (const float*)d_in[2];
    const float* desc2 = (const float*)d_in[3];
    const float* homo12 = (const float*)d_in[4];

    char* ws = (char*)d_ws;
    size_t o_wdesc = 0;
    size_t o_abf = o_wdesc + (size_t)BB * NN * CC * 4;
    size_t o_wbf = o_abf + (size_t)BB * NN * CC * 2;
    size_t o_d2bf = o_wbf + (size_t)BB * NN * CC * 2;
    size_t o_pos = o_d2bf + (size_t)BB * HWW * CC * 2;
    size_t o_excl = o_pos + (size_t)BB * NN * 4;
    size_t o_idsA = o_excl + (size_t)BB * NN * 16 * 4;
    size_t o_idsW = o_idsA + (size_t)BB * NN * 8 * 4;
    size_t o_acc = o_idsW + (size_t)BB * NN * 8 * 4;

    float* wdesc = (float*)(ws + o_wdesc);
    unsigned short* abf = (unsigned short*)(ws + o_abf);
    unsigned short* wbf = (unsigned short*)(ws + o_wbf);
    unsigned short* d2bf = (unsigned short*)(ws + o_d2bf);
    float* pos_sim = (float*)(ws + o_pos);
    int* excl = (int*)(ws + o_excl);
    int* idsA = (int*)(ws + o_idsA);
    int* idsW = (int*)(ws + o_idsW);
    float* acc = (float*)(ws + o_acc);

    init_acc_kernel<<<1, 32, 0, stream>>>(acc);
    conv_d2_kernel<<<BB * HWW, 128, 0, stream>>>(desc2, d2bf);
    prep_kernel<<<BB * NN, 32, 0, stream>>>(kp1, w_kp1, kp1_desc, desc2, homo12,
                                            wdesc, abf, wbf, pos_sim, excl);
    fos_kernel<<<BB * 64, 32, 0, stream>>>(abf, d2bf, pos_sim, excl, acc);
    topk_ids_kernel<<<BB * 64, 32, 0, stream>>>(abf, kp1, idsA);
    topk_ids_kernel<<<BB * 64, 32, 0, stream>>>(wbf, w_kp1, idsW);
    sos_kernel<<<BB * NN, 32, 0, stream>>>(kp1_desc, wdesc, idsA, idsW, acc);
    finalize_kernel<<<1, 1, 0, stream>>>(acc, (float*)d_out);
}
